// ResGConvGRU_1855425872361
// MI455X (gfx1250) — matrix operand order repaired and checked by reading, not compiled, not run on hardware
//
#include <hip/hip_runtime.h>
#include <hip/hip_bf16.h>
#include <stdint.h>

// ---------------------------------------------------------------------------
// Residual GConvGRU (ChebConv K=3, lambda_max=2.0 -> diag term == 0) for
// MI455X / gfx1250. Dense GEMMs: v_wmma_f32_16x16x32_bf16, ping-pong LDS,
// async B-tile DMA into LDS (global_load_async_to_lds_b128 + s_wait_asynccnt).
// SpMM: wave-per-edge float4 gather + global f32 atomic scatter (L2-resident).
// ---------------------------------------------------------------------------

typedef __bf16 bf16;
typedef __attribute__((ext_vector_type(16))) __bf16 bf16x16;
typedef __attribute__((ext_vector_type(8)))  __bf16 bf16x8;
typedef __attribute__((ext_vector_type(8)))  float  f32x8;
typedef __attribute__((ext_vector_type(4)))  int    v4i;

#if __has_builtin(__builtin_amdgcn_global_load_async_to_lds_b128)
#define HAVE_ASYNC_LDS 1
// Builtin signature (from clang diagnostic): param0 = int4 addrspace(1)*,
// param1 = int4 addrspace(3)*, then constant int offset, constant int cpol.
typedef __attribute__((address_space(1))) v4i glob_v4i;
typedef __attribute__((address_space(3))) v4i lds_v4i;
#else
#define HAVE_ASYNC_LDS 0
#endif

__device__ __forceinline__ void wait_async0() {
#if __has_builtin(__builtin_amdgcn_s_wait_asynccnt)
  __builtin_amdgcn_s_wait_asynccnt(0);
#else
  asm volatile("s_wait_asynccnt 0x0" ::: "memory");
#endif
}

__device__ __forceinline__ bf16 to_bf16(float f) {
  union { float f; uint32_t u; } v; v.f = f;
  uint32_t r = v.u + 0x7FFFu + ((v.u >> 16) & 1u);   // round-to-nearest-even
  uint16_t h = (uint16_t)(r >> 16);
  return __builtin_bit_cast(bf16, h);
}

__device__ __forceinline__ void atomic_fadd(float* p, float v) {
  __hip_atomic_fetch_add(p, v, __ATOMIC_RELAXED, __HIP_MEMORY_SCOPE_AGENT);
}

__device__ __forceinline__ float sigmoidf(float x) {
  return 1.0f / (1.0f + __expf(-x));
}

// ---------------------------------------------------------------------------
// Edge preprocessing
// ---------------------------------------------------------------------------
__global__ void degree_kernel(const int* __restrict__ rows,
                              const float* __restrict__ ew,
                              float* __restrict__ deg, int E) {
  int e = blockIdx.x * blockDim.x + threadIdx.x;
  if (e < E) atomic_fadd(&deg[rows[e]], ew[e]);
}

__global__ void dinv_kernel(float* __restrict__ deg, int n) {
  int i = blockIdx.x * blockDim.x + threadIdx.x;
  if (i < n) {
    float d = deg[i];
    deg[i] = (d > 0.0f) ? __frsqrt_rn(d) : 0.0f;   // in-place deg -> d^{-1/2}
  }
}

// w_e = -(2/lambda_max) * dinv[row] * ew * dinv[col]; lambda_max = 2 -> factor 1
__global__ void edgew_kernel(const int* __restrict__ rows,
                             const int* __restrict__ cols,
                             const float* __restrict__ ew,
                             const float* __restrict__ dinv,
                             float* __restrict__ w, int E) {
  int e = blockIdx.x * blockDim.x + threadIdx.x;
  if (e < E) w[e] = -dinv[rows[e]] * ew[e] * dinv[cols[e]];
}

// ---------------------------------------------------------------------------
// SpMM scatter: dst[col] += scale * w_e * src[row]   (one wave per edge,
// lane handles 4 features via float4; edge metadata is wave-uniform -> SALU)
// ---------------------------------------------------------------------------
__global__ __launch_bounds__(256)
void spmm_scatter(float* __restrict__ dst, const float* __restrict__ src,
                  const float* __restrict__ w, const int* __restrict__ rows,
                  const int* __restrict__ cols, int E, float scale) {
  long long idx = (long long)blockIdx.x * blockDim.x + threadIdx.x;
  int e = (int)(idx >> 5);
  if (e >= E) return;
  int lane = (int)(idx & 31);
  int r = rows[e];
  int c = cols[e];
  float we = w[e] * scale;
  float4 v = *((const float4*)(src + (size_t)r * 128) + lane);
  float* d = dst + (size_t)c * 128 + lane * 4;
  atomic_fadd(d + 0, we * v.x);
  atomic_fadd(d + 1, we * v.y);
  atomic_fadd(d + 2, we * v.z);
  atomic_fadd(d + 3, we * v.w);
}

// dst = -src  (init for T2 = 2*L*T1 - T0 accumulated via atomics)
__global__ void neg_init_kernel(float* __restrict__ dst,
                                const float* __restrict__ src, int n) {
  int i = blockIdx.x * blockDim.x + threadIdx.x;
  if (i < n) dst[i] = -src[i];
}

// ---------------------------------------------------------------------------
// Weight prep: W [3,128,128] f32 row-major == [384,128]; transpose to
// Wt[128][384] bf16 so each B-tile fill thread reads 32 contiguous bytes.
// ---------------------------------------------------------------------------
__global__ void weight_prep(const float* __restrict__ W, bf16* __restrict__ Wt) {
  int idx = blockIdx.x * blockDim.x + threadIdx.x;
  if (idx >= 384 * 128) return;
  int k = idx >> 7;     // 0..383
  int n = idx & 127;    // 0..127
  Wt[n * 384 + k] = to_bf16(W[idx]);
}

// ---------------------------------------------------------------------------
// WMMA GEMM: C[N,128] (+)= concat(A0|A1|A2)[N,384] @ Wt^T + bias
//   - 256 threads (8 waves as 4x2), tile 128(M) x 128(N), BK = 32
//   - ping-pong LDS; next tile's loads issued behind current tile's WMMAs
//   - B tiles DMA'd LDS-direct via global_load_async_to_lds_b128 if available
// ---------------------------------------------------------------------------
template <bool ACCUM>
__global__ __launch_bounds__(256)
void cheb_gemm(const float* __restrict__ A0, const float* __restrict__ A1,
               const float* __restrict__ A2, const bf16* __restrict__ Wt,
               const float* __restrict__ bias, float* __restrict__ C,
               int nrows) {
  __shared__ bf16 As[2][128 * 40];   // 128 rows x 32 k, stride 40 (16B pad)
  __shared__ bf16 Bs[2][128 * 40];   // 128 cols x 32 k (B transposed)

  const int tid  = threadIdx.x;
  const int lane = tid & 31;
  const int wave = tid >> 5;
  const int wm   = wave >> 1;              // 0..3 -> 32-row strip
  const int wn   = wave & 1;               // 0..1 -> 64-col strip
  const int mBase = blockIdx.x * 128;
  const int l15  = lane & 15;
  const int hiA  = lane >> 4;              // 0 or 1 (lane half)
  const int lr   = tid >> 1;               // 0..127: row (A) / col (B) fills
  const int lg   = (tid & 1) * 16;         // 0 or 16: k sub-chunk for fills

  int mrow = mBase + lr;
  if (mrow >= nrows) mrow = nrows - 1;     // clamp: keep EXEC full for WMMA

  f32x8 acc[2][4];
  const f32x8 zero = {0.f, 0.f, 0.f, 0.f, 0.f, 0.f, 0.f, 0.f};
#pragma unroll
  for (int i = 0; i < 2; ++i)
#pragma unroll
    for (int j = 0; j < 4; ++j) acc[i][j] = zero;

  union Frag { bf16x16 v; bf16x8 h[2]; };

  float4 aReg[4];
#if !HAVE_ASYNC_LDS
  bf16x8 bReg[2];
#endif

  auto issueA = [&](int kc) {
    const int seg = kc >> 2;
    const float* Aseg = (seg == 0) ? A0 : ((seg == 1) ? A1 : A2);
    const float4* ap =
        (const float4*)(Aseg + (size_t)mrow * 128 + (kc & 3) * 32 + lg);
    aReg[0] = ap[0]; aReg[1] = ap[1]; aReg[2] = ap[2]; aReg[3] = ap[3];
  };
  auto storeA = [&](int buf) {
    bf16x8 p0, p1;
    p0[0] = to_bf16(aReg[0].x); p0[1] = to_bf16(aReg[0].y);
    p0[2] = to_bf16(aReg[0].z); p0[3] = to_bf16(aReg[0].w);
    p0[4] = to_bf16(aReg[1].x); p0[5] = to_bf16(aReg[1].y);
    p0[6] = to_bf16(aReg[1].z); p0[7] = to_bf16(aReg[1].w);
    p1[0] = to_bf16(aReg[2].x); p1[1] = to_bf16(aReg[2].y);
    p1[2] = to_bf16(aReg[2].z); p1[3] = to_bf16(aReg[2].w);
    p1[4] = to_bf16(aReg[3].x); p1[5] = to_bf16(aReg[3].y);
    p1[6] = to_bf16(aReg[3].z); p1[7] = to_bf16(aReg[3].w);
    *(bf16x8*)&As[buf][lr * 40 + lg]     = p0;
    *(bf16x8*)&As[buf][lr * 40 + lg + 8] = p1;
  };

#if HAVE_ASYNC_LDS
  auto asyncB = [&](int kc, int buf) {
    const bf16* g = Wt + (size_t)lr * 384 + kc * 32 + lg;   // 32B aligned
    bf16* l = &Bs[buf][lr * 40 + lg];                       // 16B aligned
    __builtin_amdgcn_global_load_async_to_lds_b128(
        (glob_v4i*)g, (lds_v4i*)l, 0, 0);
    __builtin_amdgcn_global_load_async_to_lds_b128(
        (glob_v4i*)(g + 8), (lds_v4i*)(l + 8), 0, 0);
  };
#else
  auto issueB = [&](int kc) {
    const bf16x8* wp = (const bf16x8*)(Wt + (size_t)lr * 384 + kc * 32 + lg);
    bReg[0] = wp[0]; bReg[1] = wp[1];
  };
  auto storeB = [&](int buf) {
    *(bf16x8*)&Bs[buf][lr * 40 + lg]     = bReg[0];
    *(bf16x8*)&Bs[buf][lr * 40 + lg + 8] = bReg[1];
  };
#endif

  // ---- prologue: stage tile 0 into buffer 0 ------------------------------
  issueA(0);
#if HAVE_ASYNC_LDS
  asyncB(0, 0);
#else
  issueB(0);
  storeB(0);
#endif
  storeA(0);

#pragma unroll 1
  for (int kc = 0; kc < 12; ++kc) {
    const int cur = kc & 1;
    const int nxt = cur ^ 1;

#if HAVE_ASYNC_LDS
    wait_async0();          // our async B stores into buf[cur] are in LDS
#endif
    __syncthreads();        // publish buf[cur]

    if (kc + 1 < 12) {
      issueA(kc + 1);       // global loads overlap the WMMAs below
#if HAVE_ASYNC_LDS
      asyncB(kc + 1, nxt);  // DMA next B tile while buf[cur] is consumed
#else
      issueB(kc + 1);
#endif
    }

    // ---- fragments + WMMA (ISA 7.12.2 layouts) ---------------------------
    // A 16x32: lanes 0-15 -> K{0..7,16..23}; lanes 16-31 -> K{8..15,24..31}
    Frag af[2];
#pragma unroll
    for (int mf = 0; mf < 2; ++mf) {
      const int row = wm * 32 + mf * 16 + l15;
      const int k0  = hiA * 8;
      af[mf].h[0] = *(const bf16x8*)&As[cur][row * 40 + k0];
      af[mf].h[1] = *(const bf16x8*)&As[cur][row * 40 + k0 + 16];
    }
    // B 32x16 (transposed store): lanes 0-15 -> K{0..7,16..23}; lanes 16-31 -> K{8..15,24..31}
#pragma unroll
    for (int nf = 0; nf < 4; ++nf) {
      const int colv = wn * 64 + nf * 16 + l15;
      const int kb   = hiA * 8;
      Frag bf_;
      bf_.h[0] = *(const bf16x8*)&Bs[cur][colv * 40 + kb];
      bf_.h[1] = *(const bf16x8*)&Bs[cur][colv * 40 + kb + 16];
#pragma unroll
      for (int mf = 0; mf < 2; ++mf) {
        acc[mf][nf] = __builtin_amdgcn_wmma_f32_16x16x32_bf16(
            false, af[mf].v, false, bf_.v, (short)0, acc[mf][nf],
            false, false);
      }
    }

    if (kc + 1 < 12) {
      storeA(nxt);          // ds_store into the other buffer: no race
#if !HAVE_ASYNC_LDS
      storeB(nxt);
#endif
    }
  }

  // ---- epilogue: C/D layout lanes 0-15: M=v, N=lane; 16-31: M=v+8 --------
  const int moff = hiA * 8;
  const int col0 = wn * 64 + l15;
  const bool fullTile = (mBase + 128 <= nrows);   // block-uniform branch
  if (fullTile) {
#pragma unroll
    for (int mf = 0; mf < 2; ++mf) {
#pragma unroll
      for (int nf = 0; nf < 4; ++nf) {
        const int col = col0 + nf * 16;
        const float bcol = bias[col];
        const int mrow0 = mBase + wm * 32 + mf * 16 + moff;
#pragma unroll
        for (int v = 0; v < 8; ++v) {
          const size_t off = (size_t)(mrow0 + v) * 128 + col;
          float val = acc[mf][nf][v] + bcol;
          if (ACCUM) val += C[off];
          C[off] = val;
        }
      }
    }
  } else {
#pragma unroll
    for (int mf = 0; mf < 2; ++mf) {
#pragma unroll
      for (int nf = 0; nf < 4; ++nf) {
        const int col = col0 + nf * 16;
        const float bcol = bias[col];
#pragma unroll
        for (int v = 0; v < 8; ++v) {
          const int m = mBase + wm * 32 + mf * 16 + moff + v;
          if (m < nrows) {
            const size_t off = (size_t)m * 128 + col;
            float val = acc[mf][nf][v] + bcol;
            if (ACCUM) val += C[off];
            C[off] = val;
          }
        }
      }
    }
  }
}

// ---------------------------------------------------------------------------
// Elementwise GRU pieces
// ---------------------------------------------------------------------------
__global__ void hr_kernel(const float* __restrict__ H,
                          const float* __restrict__ accR,
                          float* __restrict__ HR, int n) {
  int i = blockIdx.x * blockDim.x + threadIdx.x;
  if (i < n) HR[i] = H[i] * sigmoidf(accR[i]);
}

__global__ void final_kernel(const float* __restrict__ H,
                             const float* __restrict__ accZ,
                             const float* __restrict__ accS,
                             float* __restrict__ out, int n) {
  int i = blockIdx.x * blockDim.x + threadIdx.x;
  if (i < n) {
    float z  = sigmoidf(accZ[i]);
    float s  = accS[i];
    float ht = tanhf(s) + s;          // residual candidate
    out[i] = z * H[i] + (1.0f - z) * ht;
  }
}

// ---------------------------------------------------------------------------
// Host-side orchestration
// ---------------------------------------------------------------------------
static inline int cdiv(long long a, long long b) { return (int)((a + b - 1) / b); }

extern "C" void kernel_launch(void* const* d_in, const int* in_sizes, int n_in,
                              void* d_out, int out_size, void* d_ws, size_t ws_size,
                              hipStream_t stream) {
  const float* X   = (const float*)d_in[0];
  const int*   ei  = (const int*)d_in[1];
  const float* ew  = (const float*)d_in[2];
  const float* H   = (const float*)d_in[3];
  const float* Wxz = (const float*)d_in[4];  const float* bxz = (const float*)d_in[5];
  const float* Whz = (const float*)d_in[6];  const float* bhz = (const float*)d_in[7];
  const float* Wxr = (const float*)d_in[8];  const float* bxr = (const float*)d_in[9];
  const float* Whr = (const float*)d_in[10]; const float* bhr = (const float*)d_in[11];
  const float* Wxh = (const float*)d_in[12]; const float* bxh = (const float*)d_in[13];
  const float* Whh = (const float*)d_in[14]; const float* bhh = (const float*)d_in[15];

  const int Nn = in_sizes[0] / 128;     // 50000
  const int E  = in_sizes[2];           // 600000
  const int* rows = ei;
  const int* cols = ei + E;
  const long long NF = (long long)Nn * 128;

  // ---- workspace carving -------------------------------------------------
  char* p = (char*)d_ws;
  auto alloc = [&](size_t bytes) -> void* {
    void* r = (void*)p;
    p += (bytes + 255) & ~(size_t)255;
    return r;
  };
  float* deg   = (float*)alloc((size_t)Nn * 4);        // reused as dinv
  float* wbuf  = (float*)alloc((size_t)E * 4);
  float* TX1   = (float*)alloc((size_t)NF * 4);
  float* TX2   = (float*)alloc((size_t)NF * 4);
  float* TH1   = (float*)alloc((size_t)NF * 4);        // later reused as THR1
  float* TH2   = (float*)alloc((size_t)NF * 4);        // later reused as THR2
  float* HR    = (float*)alloc((size_t)NF * 4);
  float* accZ  = (float*)alloc((size_t)NF * 4);
  float* accR  = (float*)alloc((size_t)NF * 4);
  float* accS  = (float*)alloc((size_t)NF * 4);
  bf16* WxzT = (bf16*)alloc((size_t)384 * 128 * 2);
  bf16* WhzT = (bf16*)alloc((size_t)384 * 128 * 2);
  bf16* WxrT = (bf16*)alloc((size_t)384 * 128 * 2);
  bf16* WhrT = (bf16*)alloc((size_t)384 * 128 * 2);
  bf16* WxhT = (bf16*)alloc((size_t)384 * 128 * 2);
  bf16* WhhT = (bf16*)alloc((size_t)384 * 128 * 2);

  const int TB = 256;
  const int gE   = cdiv(E, TB);
  const int gN   = cdiv(Nn, TB);
  const int gNF  = cdiv(NF, TB);
  const int gSp  = cdiv((long long)E * 32, TB);
  const int gW   = cdiv(384 * 128, TB);
  const int gGemm = cdiv(Nn, 128);
  float* out = (float*)d_out;

  // ---- edge normalization ------------------------------------------------
  (void)hipMemsetAsync(deg, 0, (size_t)Nn * 4, stream);
  degree_kernel<<<gE, TB, 0, stream>>>(rows, ew, deg, E);
  dinv_kernel<<<gN, TB, 0, stream>>>(deg, Nn);
  edgew_kernel<<<gE, TB, 0, stream>>>(rows, cols, ew, deg, wbuf, E);

  // ---- weight transpose/convert (f32 -> bf16, [384,128] -> [128][384]) ---
  weight_prep<<<gW, TB, 0, stream>>>(Wxz, WxzT);
  weight_prep<<<gW, TB, 0, stream>>>(Whz, WhzT);
  weight_prep<<<gW, TB, 0, stream>>>(Wxr, WxrT);
  weight_prep<<<gW, TB, 0, stream>>>(Whr, WhrT);
  weight_prep<<<gW, TB, 0, stream>>>(Wxh, WxhT);
  weight_prep<<<gW, TB, 0, stream>>>(Whh, WhhT);

  // ---- Chebyshev bases (diag term is exactly 0 for lambda_max = 2) -------
  // X basis: TX1 = L X ; TX2 = 2 L TX1 - X
  (void)hipMemsetAsync(TX1, 0, (size_t)NF * 4, stream);
  spmm_scatter<<<gSp, TB, 0, stream>>>(TX1, X, wbuf, rows, cols, E, 1.0f);
  neg_init_kernel<<<gNF, TB, 0, stream>>>(TX2, X, (int)NF);
  spmm_scatter<<<gSp, TB, 0, stream>>>(TX2, TX1, wbuf, rows, cols, E, 2.0f);
  // H basis
  (void)hipMemsetAsync(TH1, 0, (size_t)NF * 4, stream);
  spmm_scatter<<<gSp, TB, 0, stream>>>(TH1, H, wbuf, rows, cols, E, 1.0f);
  neg_init_kernel<<<gNF, TB, 0, stream>>>(TH2, H, (int)NF);
  spmm_scatter<<<gSp, TB, 0, stream>>>(TH2, TH1, wbuf, rows, cols, E, 2.0f);

  // ---- gates: Z and R (each = X-basis GEMM, then accumulate H-basis) -----
  cheb_gemm<false><<<gGemm, TB, 0, stream>>>(X, TX1, TX2, WxzT, bxz, accZ, Nn);
  cheb_gemm<true ><<<gGemm, TB, 0, stream>>>(H, TH1, TH2, WhzT, bhz, accZ, Nn);
  cheb_gemm<false><<<gGemm, TB, 0, stream>>>(X, TX1, TX2, WxrT, bxr, accR, Nn);
  cheb_gemm<true ><<<gGemm, TB, 0, stream>>>(H, TH1, TH2, WhrT, bhr, accR, Nn);

  // ---- HR = H * sigmoid(accR); its basis reuses TH1/TH2 buffers ----------
  hr_kernel<<<gNF, TB, 0, stream>>>(H, accR, HR, (int)NF);
  (void)hipMemsetAsync(TH1, 0, (size_t)NF * 4, stream);
  spmm_scatter<<<gSp, TB, 0, stream>>>(TH1, HR, wbuf, rows, cols, E, 1.0f);
  neg_init_kernel<<<gNF, TB, 0, stream>>>(TH2, HR, (int)NF);
  spmm_scatter<<<gSp, TB, 0, stream>>>(TH2, TH1, wbuf, rows, cols, E, 2.0f);

  // ---- candidate pre-activation s ----------------------------------------
  cheb_gemm<false><<<gGemm, TB, 0, stream>>>(X,  TX1, TX2, WxhT, bxh, accS, Nn);
  cheb_gemm<true ><<<gGemm, TB, 0, stream>>>(HR, TH1, TH2, WhhT, bhh, accS, Nn);

  // ---- H' = Z*H + (1-Z)*(tanh(s)+s) --------------------------------------
  final_kernel<<<gNF, TB, 0, stream>>>(H, accZ, accS, out, (int)NF);
}
